// VanillaMasker_88957362635160
// MI455X (gfx1250) — matrix operand order repaired and checked by reading, not compiled, not run on hardware
//
#include <hip/hip_runtime.h>
#include <hip/hip_bf16.h>

typedef __attribute__((ext_vector_type(16))) __bf16 v16bf;
typedef __attribute__((ext_vector_type(8)))  __bf16 v8bf;
typedef __attribute__((ext_vector_type(8)))  float  v8f;
typedef __attribute__((ext_vector_type(2)))  float  v2f;

union BF16x16 { v16bf v; v8bf h[2]; };

// ---- problem constants ----
#define NB   32        // batch
#define CDIM 768
#define LTOK 1024
#define TOPK 512
#define CBD  32        // codebook dim

// ---- output float offsets (tuple concatenated flat) ----
#define O_SAMPLED 0          // 32*32*512      = 524288
#define O_TOPK    524288     // 32*512         = 16384
#define O_REMAIN  540672     // 32*512         = 16384
#define O_BMAP    557056     // 32*512*512     = 8388608
#define O_SMAP    8945664    // 32*512*512     = 8388608
#define O_SQMASK  17334272   // 32*1024        = 32768
#define O_SCORE   17367040   // 32*512         = 16384

// LDS row strides (padded: stride%64 dwords != 0 -> conflict-free row access)
#define ASTR 776   // bf16 elements per row (1552B, 16B aligned)
#define FSTR 772   // f32 elements per row  (3088B, 8B aligned)

// ---------------------------------------------------------------------------
// Prep: split w1 into bf16 hi/lo (hi = rne(v), lo = rne(v - hi)); init min/max
// ---------------------------------------------------------------------------
__global__ __launch_bounds__(256)
void prep_kernel(const float* __restrict__ w1,
                 __bf16* __restrict__ w1hi, __bf16* __restrict__ w1lo,
                 unsigned* __restrict__ mm) {
  int i = blockIdx.x * 256 + threadIdx.x;
  if (i == 0) { mm[0] = 0x7F800000u; mm[1] = 0u; }
  if (i < CDIM * CDIM) {
    float v = w1[i];
    __bf16 h = (__bf16)v;
    w1hi[i] = h;
    w1lo[i] = (__bf16)(v - (float)h);
  }
}

// ---------------------------------------------------------------------------
// Score: fused  sigmoid( relu(x*W1^T + b1) . w2 + b2 )  per token.
// Block = 128 threads (4 waves). Block owns a 16-token M-tile; each wave owns
// 12 N-tiles of 16; K looped 768 in steps of 32 with split-bf16 WMMA.
// ---------------------------------------------------------------------------
__global__ __launch_bounds__(128)
void score_kernel(const float* __restrict__ img,
                  const __bf16* __restrict__ w1hi, const __bf16* __restrict__ w1lo,
                  const float* __restrict__ b1, const float* __restrict__ w2,
                  const float* __restrict__ b2,
                  float* __restrict__ pred, unsigned* __restrict__ mm) {
  __shared__ __bf16 Ahi[16 * ASTR];
  __shared__ __bf16 Alo[16 * ASTR];
  __shared__ float  Psum[64];

  const int tid  = threadIdx.x;
  const int lane = tid & 31;
  const int wave = tid >> 5;
  const int tile = blockIdx.x;            // 2048 tiles
  const int b    = tile >> 6;             // 64 tiles per batch
  const int l0   = (tile & 63) << 4;
  const float* xb = img + (size_t)b * CDIM * LTOK + l0;

  // Stage x tile -> LDS as bf16 hi/lo.  i&15 = token (coalesced l), i>>4 = ch.
  for (int i = tid; i < 16 * CDIM; i += 128) {
    int m = i & 15, c = i >> 4;
    float v = xb[(size_t)c * LTOK + m];
    __bf16 h = (__bf16)v;
    Ahi[m * ASTR + c] = h;
    Alo[m * ASTR + c] = (__bf16)(v - (float)h);
  }
  __syncthreads();

  const int mcol = lane & 15;   // A row / B column / C column
  const int half = lane >> 4;
  const __bf16* aHiP = &Ahi[mcol * ASTR + half * 8];
  const __bf16* aLoP = &Alo[mcol * ASTR + half * 8];

  float rowsum[8] = {0.f,0.f,0.f,0.f,0.f,0.f,0.f,0.f};

  for (int nt = 0; nt < 12; ++nt) {
    const int n = ((wave * 12 + nt) << 4) + mcol;
    const __bf16* bHiP = w1hi + (size_t)n * CDIM + half * 8;
    const __bf16* bLoP = w1lo + (size_t)n * CDIM + half * 8;
    v8f acc = {};
    for (int kb = 0; kb < CDIM; kb += 32) {
      BF16x16 ah, al, bh, bl;
      ah.h[0] = *(const v8bf*)(aHiP + kb);      // K = kb+8h .. +7
      ah.h[1] = *(const v8bf*)(aHiP + kb + 16); // K = kb+16+8h .. +7
      al.h[0] = *(const v8bf*)(aLoP + kb);
      al.h[1] = *(const v8bf*)(aLoP + kb + 16);
      bh.h[0] = *(const v8bf*)(bHiP + kb);      // K = kb+8h .. +7
      bh.h[1] = *(const v8bf*)(bHiP + kb + 16); // K = kb+16+8h .. +7
      bl.h[0] = *(const v8bf*)(bLoP + kb);
      bl.h[1] = *(const v8bf*)(bLoP + kb + 16);
      acc = __builtin_amdgcn_wmma_f32_16x16x32_bf16(false, ah.v, false, bh.v, (short)0, acc, false, false);
      acc = __builtin_amdgcn_wmma_f32_16x16x32_bf16(false, ah.v, false, bl.v, (short)0, acc, false, false);
      acc = __builtin_amdgcn_wmma_f32_16x16x32_bf16(false, al.v, false, bh.v, (short)0, acc, false, false);
    }
    const float bias = b1[n], wv = w2[n];
#pragma unroll
    for (int i = 0; i < 8; ++i) {
      float hx = acc[i] + bias;                 // row i + 8*half, col n
      hx = hx > 0.f ? hx : 0.f;
      rowsum[i] += hx * wv;
    }
  }

  // Reduce over the 16 columns held by each lane-half.
#pragma unroll
  for (int msk = 1; msk < 16; msk <<= 1)
#pragma unroll
    for (int i = 0; i < 8; ++i)
      rowsum[i] += __shfl_xor(rowsum[i], msk, 32);
  if (mcol == 0) {
#pragma unroll
    for (int i = 0; i < 8; ++i)
      Psum[wave * 16 + half * 8 + i] = rowsum[i];
  }
  __syncthreads();

  if (tid < 16) {
    float t = Psum[tid] + Psum[16 + tid] + Psum[32 + tid] + Psum[48 + tid] + b2[0];
    float s = 1.f / (1.f + __expf(-t));
    pred[b * LTOK + l0 + tid] = s;
    unsigned u = __float_as_uint(s);            // sigmoid>0: uint order == float order
    atomicMin(&mm[0], u);
    atomicMax(&mm[1], u);
  }
}

// ---------------------------------------------------------------------------
// Per-batch descending bitonic sort of 1024 scores; emits topk/remain/score,
// binary mask (ws + squeezed_mask output), and topk indices for the gather.
// ---------------------------------------------------------------------------
__global__ __launch_bounds__(512)
void sort_kernel(const float* __restrict__ pred, float* __restrict__ out,
                 int* __restrict__ tidx, float* __restrict__ mask) {
  __shared__ float sk[LTOK];
  __shared__ int   si[LTOK];
  const int b = blockIdx.x, t = threadIdx.x;
  sk[t]       = pred[b * LTOK + t];       si[t]       = t;
  sk[t + 512] = pred[b * LTOK + t + 512]; si[t + 512] = t + 512;

  for (int k = 2; k <= LTOK; k <<= 1) {
    for (int j = k >> 1; j > 0; j >>= 1) {
      __syncthreads();
      int i = 2 * t - (t & (j - 1));
      int ixj = i + j;
      bool up = (i & k) == 0;               // descending segments
      float a = sk[i], c = sk[ixj];
      if (up ? (a < c) : (a > c)) {
        sk[i] = c; sk[ixj] = a;
        int tmp = si[i]; si[i] = si[ixj]; si[ixj] = tmp;
      }
    }
  }
  __syncthreads();

  for (int r = t; r < LTOK; r += 512) {
    int idx = si[r];
    float mv = (r < TOPK) ? 1.f : 0.f;
    mask[b * LTOK + idx] = mv;
    out[O_SQMASK + b * LTOK + idx] = mv;
    if (r < TOPK) {
      out[O_TOPK  + b * TOPK + r] = (float)idx;
      out[O_SCORE + b * TOPK + r] = sk[r];
      tidx[b * TOPK + r] = idx;
    } else {
      out[O_REMAIN + b * TOPK + (r - TOPK)] = (float)idx;
    }
  }
}

// ---------------------------------------------------------------------------
// Gather + layernorm + score-scale + projection to 32 dims via f32 WMMA.
// One wave per block; block owns 16 gathered tokens, N = 32 (two 16-tiles).
// (x-mu)*rstd*score folded into the A-operand fetch from LDS.
// ---------------------------------------------------------------------------
__global__ __launch_bounds__(32)
void project_kernel(const float* __restrict__ img, const float* __restrict__ wproj,
                    const float* __restrict__ pred, const int* __restrict__ tidx,
                    float* __restrict__ out) {
  __shared__ float A[16 * FSTR];
  __shared__ int   Sidx[16];
  __shared__ float Smu[16], Sf[16];

  const int lane = threadIdx.x;
  const int blk  = blockIdx.x;           // 32 * 32 blocks
  const int b    = blk >> 5;
  const int k0   = (blk & 31) << 4;      // offset into topk list
  const float* xb = img + (size_t)b * CDIM * LTOK;

  if (lane < 16) Sidx[lane] = tidx[b * TOPK + k0 + lane];
  __syncthreads();

  for (int i = lane; i < 16 * CDIM; i += 32) {
    int m = i & 15, c = i >> 4;
    A[m * FSTR + c] = xb[(size_t)c * LTOK + Sidx[m]];
  }
  __syncthreads();

  {
    int m = lane & 15, h = lane >> 4;
    float s = 0.f, sq = 0.f;
    for (int c = h * 384; c < h * 384 + 384; ++c) {
      float v = A[m * FSTR + c];
      s += v; sq += v * v;
    }
    s  += __shfl_xor(s, 16, 32);
    sq += __shfl_xor(sq, 16, 32);
    if (h == 0) {
      float mu  = s * (1.f / 768.f);
      float var = sq * (1.f / 768.f) - mu * mu;
      Smu[m] = mu;
      Sf[m]  = rsqrtf(var + 1e-5f) * pred[b * LTOK + Sidx[m]];
    }
  }
  __syncthreads();

  const int m = lane & 15, h = lane >> 4;
  const float mu = Smu[m], f = Sf[m];
  const float* aP  = &A[m * FSTR + (h << 1)];
  const float* b0P = wproj + (size_t)m * CDIM + (h << 1);          // n = m
  const float* b1P = wproj + (size_t)(m + 16) * CDIM + (h << 1);   // n = m+16
  v8f acc0 = {}, acc1 = {};
  for (int kb = 0; kb < CDIM; kb += 4) {
    float2 raw = *(const float2*)(aP + kb);
    v2f a; a.x = (raw.x - mu) * f; a.y = (raw.y - mu) * f;
    v2f bv0 = *(const v2f*)(b0P + kb);
    v2f bv1 = *(const v2f*)(b1P + kb);
    acc0 = __builtin_amdgcn_wmma_f32_16x16x4_f32(false, a, false, bv0, (short)0, acc0, false, false);
    acc1 = __builtin_amdgcn_wmma_f32_16x16x4_f32(false, a, false, bv1, (short)0, acc1, false, false);
  }
  // sampled layout: (B, 32, 512) -> out[b*16384 + n*512 + (k0+row)]
  float* ob = out + O_SAMPLED + b * (CBD * TOPK) + k0;
#pragma unroll
  for (int i = 0; i < 8; ++i) {
    int row = i + h * 8;
    ob[m * TOPK + row]        = acc0[i];
    ob[(m + 16) * TOPK + row] = acc1[i];
  }
}

// ---------------------------------------------------------------------------
// binary_map + score_map: nearest upsample x16 of mask / normalized scores.
// ---------------------------------------------------------------------------
__global__ __launch_bounds__(256)
void maps_kernel(const float* __restrict__ pred, const float* __restrict__ mask,
                 const unsigned* __restrict__ mm, float* __restrict__ out) {
  int id = blockIdx.x * 256 + threadIdx.x;         // 32*512*512
  if (id >= NB * 512 * 512) return;
  int b = id >> 18;
  int rem = id & 262143;
  int y = rem >> 9, x = rem & 511;
  int l = ((y >> 4) << 5) + (x >> 4);
  float mn = __uint_as_float(mm[0]);
  float mx = __uint_as_float(mm[1]);
  float inv = 1.f / fmaxf(mx - mn, 1e-5f);
  out[O_BMAP + id] = mask[b * LTOK + l];
  out[O_SMAP + id] = (pred[b * LTOK + l] - mn) * inv;
}

// ---------------------------------------------------------------------------
extern "C" void kernel_launch(void* const* d_in, const int* in_sizes, int n_in,
                              void* d_out, int out_size, void* d_ws, size_t ws_size,
                              hipStream_t stream) {
  const float* img = (const float*)d_in[0];
  const float* w1  = (const float*)d_in[1];
  const float* b1  = (const float*)d_in[2];
  const float* w2  = (const float*)d_in[3];
  const float* b2  = (const float*)d_in[4];
  const float* wpj = (const float*)d_in[5];
  float* out = (float*)d_out;
  char*  ws  = (char*)d_ws;

  float*    pred = (float*)(ws + 0);          // 32768 f32
  float*    mask = (float*)(ws + 131072);     // 32768 f32
  int*      tidx = (int*)  (ws + 262144);     // 16384 i32
  unsigned* mm   = (unsigned*)(ws + 327680);  // min/max bits
  __bf16*   w1hi = (__bf16*)(ws + 327936);    // 768*768 bf16
  __bf16*   w1lo = (__bf16*)(ws + 1507584);   // 768*768 bf16

  prep_kernel   <<<2304, 256, 0, stream>>>(w1, w1hi, w1lo, mm);
  score_kernel  <<<2048, 128, 0, stream>>>(img, w1hi, w1lo, b1, w2, b2, pred, mm);
  sort_kernel   <<<NB, 512, 0, stream>>>(pred, out, tidx, mask);
  project_kernel<<<NB * 32, 32, 0, stream>>>(img, wpj, pred, tidx, out);
  maps_kernel   <<<(NB * 512 * 512) / 256, 256, 0, stream>>>(pred, mask, mm, out);
}
